// StructPool_43104291783361
// MI455X (gfx1250) — matrix operand order repaired and checked by reading, not compiled, not run on hardware
//
#include <hip/hip_runtime.h>
#include <hip/hip_bf16.h>
#include <math.h>

// ---------------- problem constants (from reference) ----------------
#define N_NODES 200000
#define DIM     256
#define DH      128          // D/2
#define N_EDGES 6400000
#define K_TOP   100000       // RATIO*N
#define M_SORT  262144       // next pow2 >= N
#define EPB     2048         // edges per block (256 thr * 8)
#define NBLK_E  ((N_EDGES + EPB - 1) / EPB)
#define NEG_SLOPE 0.2f
#define STR     (DIM + 8)    // LDS row stride in halves (16B-aligned rows)
#define ROWS    32           // rows per block: each B fragment feeds 2 WMMAs

typedef _Float16 v16h __attribute__((ext_vector_type(16)));
typedef _Float16 v8h  __attribute__((ext_vector_type(8)));
typedef _Float16 v4h  __attribute__((ext_vector_type(4)));
typedef float    v8f  __attribute__((ext_vector_type(8)));

__device__ __forceinline__ float lrelu(float v) { return v > 0.f ? v : NEG_SLOPE * v; }

// A fragment (16x32 f16, MxK) from LDS tile; ISA 7.12.2 wave32 layout.
// Per lane: two contiguous 8-half runs -> 2x ds_load_b128.
__device__ __forceinline__ v16h load_a_lds(const _Float16* xs, int lane, int kb) {
  int m   = lane & 15;
  int khi = (lane < 16) ? 0 : 8;
  const _Float16* p = xs + m * STR + kb + khi;
  v8h lo = *(const v8h*)p;          // K = kb+khi+0..7
  v8h hi = *(const v8h*)(p + 16);   // K = kb+khi+16..23
  v16h a;
#pragma unroll
  for (int i = 0; i < 8; ++i) { a[i] = lo[i]; a[i + 8] = hi[i]; }
  return a;
}

// B fragment from fragment-major packed weights: [tile][lane][16 halves]
// (one aligned 32B load -> 2x global_load_b128, L2-resident)
__device__ __forceinline__ v16h load_b_pk(const _Float16* __restrict__ wpk, int tile, int lane) {
  return *(const v16h*)(wpk + ((size_t)tile * 32 + lane) * 16);
}

// x rows -> f16 LDS tile: float4 global loads + 8B LDS stores (vectorized)
__device__ __forceinline__ void stage_row4(_Float16* xs, const float* __restrict__ x,
                                           int m, int c, size_t row) {
  const float4 f = *(const float4*)(x + row * DIM + c);
  v4h h; h[0] = (_Float16)f.x; h[1] = (_Float16)f.y; h[2] = (_Float16)f.z; h[3] = (_Float16)f.w;
  *(v4h*)(xs + m * STR + c) = h;
}

// ---------------- setup kernels ----------------
__global__ void init_kernel(float* deg, float* sumsq, int* nmap) {
  int i = blockIdx.x * blockDim.x + threadIdx.x;
  if (i < N_NODES) { deg[i] = 0.f; nmap[i] = -1; }
  if (i == 0) sumsq[0] = 0.f;
}

// Pack f32 row-major weights [256][ncols] into f16 WMMA B-fragment order:
// dst[((ct*8+kq)*32 + lane)*16 + e] = src[(kq*32 + (lane<16?0:8) + (e<8?e:e+8))][ct*16 + (lane&15)]
__global__ void pack_b(const float* __restrict__ src, _Float16* __restrict__ dst,
                       int nct, int ncols) {
  int total = nct * 8 * 32 * 16;
  for (int i = blockIdx.x * blockDim.x + threadIdx.x; i < total;
       i += gridDim.x * blockDim.x) {
    int e    = i & 15;
    int lane = (i >> 4) & 31;
    int tile = i >> 9;
    int kq   = tile & 7;
    int ct   = tile >> 3;
    int k    = kq * 32 + ((lane < 16) ? 0 : 8) + ((e < 8) ? e : e + 8);
    int n    = ct * 16 + (lane & 15);
    dst[i] = (_Float16)src[(size_t)k * ncols + n];
  }
}

// rank-1 term: csum[j] = sum_i w_s1[256+i][j]  (broadcast half of struct_in)
__global__ void csum_kernel(const float* __restrict__ ws1, float* __restrict__ csum) {
  int j = threadIdx.x;          // one block of 256
  float s = 0.f;
  for (int i = 0; i < DIM; ++i) s += ws1[(size_t)(DIM + i) * DIM + j];
  csum[j] = s;
}

__global__ void degree_kernel(const long long* __restrict__ ei, float* __restrict__ deg) {
  long long i = (long long)blockIdx.x * blockDim.x + threadIdx.x;
  if (i < N_EDGES) atomicAdd(&deg[ei[i]], 1.0f);
}

__global__ void sumsq_kernel(const float* __restrict__ deg, float* __restrict__ sumsq) {
  __shared__ float sh[256];
  int i = blockIdx.x * 256 + threadIdx.x;
  float v = (i < N_NODES) ? deg[i] : 0.f;
  sh[threadIdx.x] = v * v;
  __syncthreads();
  for (int s = 128; s > 0; s >>= 1) {
    if (threadIdx.x < s) sh[threadIdx.x] += sh[threadIdx.x + s];
    __syncthreads();
  }
  if (threadIdx.x == 0) atomicAdd(sumsq, sh[0]);
}

// ---------------- fused score kernel: both MLP branches via WMMA ----------------
// 32 rows/block: each B fragment is reused by two row-tiles (halves L2 B-traffic).
// Per-lane partials go to LDS (plain stores); two-stage tree reduce.
__global__ __launch_bounds__(256)
void scores_kernel(const float* __restrict__ x,
                   const _Float16* __restrict__ ws1p, const float* __restrict__ bs1,
                   const float* __restrict__ ws2, const float* __restrict__ bs2,
                   const _Float16* __restrict__ wf1p, const float* __restrict__ bf1,
                   const float* __restrict__ wf2, const float* __restrict__ bf2,
                   const float* __restrict__ csum,
                   const float* __restrict__ deg, const float* __restrict__ sumsq,
                   const float* __restrict__ alpha, const float* __restrict__ beta,
                   float* __restrict__ scores) {
  __shared__ __align__(32) _Float16 xs[ROWS * STR];
  __shared__ float hs[ROWS][DIM + DH];    // weighted+activated partials (struct | feat)
  __shared__ float part[ROWS][8];
  __shared__ float dn[ROWS];
  int tid = threadIdx.x, lane = tid & 31, wave = tid >> 5;
  int r0 = blockIdx.x * ROWS;

  __builtin_prefetch(ws1p, 0, 3);   // global_prefetch_b8: warm weight lines
  for (int i = tid; i < ROWS * (DIM / 4); i += 256) {
    int m = i >> 6, c = (i & 63) * 4;
    stage_row4(xs, x, m, c, (size_t)(r0 + m));
  }
  if (tid < ROWS) {
    float invn = 1.0f / fmaxf(sqrtf(sumsq[0]), 1e-12f);
    dn[tid] = deg[r0 + tid] * invn;
  }
  __syncthreads();

  float sa = 1.f / (1.f + __expf(-alpha[0]));
  float sb = 1.f / (1.f + __expf(-beta[0]));
  float wa = sa / (sa + sb), wb = sb / (sa + sb);

  // struct branch: 16 column tiles over 8 waves, 2 row-tiles per B fragment
  for (int ct = wave; ct < 16; ct += 8) {
    v8f acc0 = {}, acc1 = {};
#pragma unroll
    for (int kq = 0; kq < 8; ++kq) {
      v16h b  = load_b_pk(ws1p, ct * 8 + kq, lane);
      v16h a0 = load_a_lds(xs,            lane, kq * 32);
      v16h a1 = load_a_lds(xs + 16 * STR, lane, kq * 32);
      acc0 = __builtin_amdgcn_wmma_f32_16x16x32_f16(false, a0, false, b, (short)0, acc0,
                                                    false, false);
      acc1 = __builtin_amdgcn_wmma_f32_16x16x32_f16(false, a1, false, b, (short)0, acc1,
                                                    false, false);
    }
    int mg = (lane < 16) ? 0 : 8;
    int n  = ct * 16 + (lane & 15);
    float w2 = ws2[n] * wa, bb = bs1[n], cs = csum[n];
#pragma unroll
    for (int v = 0; v < 8; ++v) {
      int m0 = v + mg, m1 = m0 + 16;
      hs[m0][n] = lrelu(acc0[v] + bb + dn[m0] * cs) * w2;  // plain ds_store_b32
      hs[m1][n] = lrelu(acc1[v] + bb + dn[m1] * cs) * w2;
    }
  }

  // feature branch: 8 column tiles, one per wave
  {
    int ct = wave;
    v8f acc0 = {}, acc1 = {};
#pragma unroll
    for (int kq = 0; kq < 8; ++kq) {
      v16h b  = load_b_pk(wf1p, ct * 8 + kq, lane);
      v16h a0 = load_a_lds(xs,            lane, kq * 32);
      v16h a1 = load_a_lds(xs + 16 * STR, lane, kq * 32);
      acc0 = __builtin_amdgcn_wmma_f32_16x16x32_f16(false, a0, false, b, (short)0, acc0,
                                                    false, false);
      acc1 = __builtin_amdgcn_wmma_f32_16x16x32_f16(false, a1, false, b, (short)0, acc1,
                                                    false, false);
    }
    int mg = (lane < 16) ? 0 : 8;
    int n  = ct * 16 + (lane & 15);
    float w2 = wf2[n] * wb, bb = bf1[n];
#pragma unroll
    for (int v = 0; v < 8; ++v) {
      hs[v + mg][DIM + n]      = lrelu(acc0[v] + bb) * w2;
      hs[v + mg + 16][DIM + n] = lrelu(acc1[v] + bb) * w2;
    }
  }
  __syncthreads();

  // two-stage tree reduce: 384 cols -> 8 segments of 48 -> score
  {
    int m = tid >> 3, seg = tid & 7;   // 256 threads: (row 0..31, segment 0..7)
    float s = 0.f;
#pragma unroll
    for (int j = 0; j < (DIM + DH) / 8; ++j) s += hs[m][seg * ((DIM + DH) / 8) + j];
    part[m][seg] = s;
  }
  __syncthreads();
  if (tid < ROWS) {
    float t = 0.f;
#pragma unroll
    for (int j = 0; j < 8; ++j) t += part[tid][j];
    scores[r0 + tid] = t + wa * bs2[0] + wb * bf2[0];
  }
}

// ---------------- top-k via bitonic sort on L2-resident keys ----------------
__device__ __forceinline__ unsigned fkey(float f) {  // ascending key <-> ascending float
  unsigned u = __float_as_uint(f);
  return (u & 0x80000000u) ? ~u : (u | 0x80000000u);
}

__global__ void build_keys(const float* __restrict__ scores, unsigned long long* __restrict__ keys) {
  int i = blockIdx.x * 256 + threadIdx.x;
  if (i < M_SORT)
    keys[i] = (i < N_NODES)
                  ? (((unsigned long long)(~fkey(scores[i])) << 32) | (unsigned)i)
                  : 0xFFFFFFFFFFFFFFFFull;   // pad sorts last; ties break by low index
}

__global__ void bitonic_step(unsigned long long* __restrict__ d, unsigned j, unsigned kk) {
  unsigned i   = blockIdx.x * 256 + threadIdx.x;
  unsigned ixj = i ^ j;
  if (ixj > i) {
    unsigned long long a = d[i], b = d[ixj];
    bool asc = ((i & kk) == 0);
    if ((a > b) == asc) { d[i] = b; d[ixj] = a; }
  }
}

__global__ void extract_topk(const unsigned long long* __restrict__ keys,
                             const float* __restrict__ scores,
                             int* __restrict__ topk, int* __restrict__ nmap,
                             float* __restrict__ out_scores) {
  int i = blockIdx.x * 256 + threadIdx.x;
  if (i < K_TOP) {
    int idx = (int)(unsigned)(keys[i] & 0xFFFFFFFFull);
    topk[i]       = idx;
    nmap[idx]     = i;
    out_scores[i] = scores[idx];
  }
}

// ---------------- pooled GEMM: lrelu(x[topk] @ w_p + b_p) ----------------
__global__ __launch_bounds__(256)
void pool_kernel(const float* __restrict__ x, const int* __restrict__ topk,
                 const _Float16* __restrict__ wpp, const float* __restrict__ bp,
                 float* __restrict__ out) {
  __shared__ __align__(32) _Float16 xs[ROWS * STR];
  __shared__ int gidx[ROWS];
  int tid = threadIdx.x, lane = tid & 31, wave = tid >> 5;
  int r0 = blockIdx.x * ROWS;
  if (tid < ROWS) gidx[tid] = topk[r0 + tid];
  __syncthreads();
  for (int i = tid; i < ROWS * (DIM / 4); i += 256) {
    int m = i >> 6, c = (i & 63) * 4;
    stage_row4(xs, x, m, c, (size_t)gidx[m]);
  }
  __syncthreads();
  for (int ct = wave; ct < 16; ct += 8) {
    v8f acc0 = {}, acc1 = {};
#pragma unroll
    for (int kq = 0; kq < 8; ++kq) {
      v16h b  = load_b_pk(wpp, ct * 8 + kq, lane);
      v16h a0 = load_a_lds(xs,            lane, kq * 32);
      v16h a1 = load_a_lds(xs + 16 * STR, lane, kq * 32);
      acc0 = __builtin_amdgcn_wmma_f32_16x16x32_f16(false, a0, false, b, (short)0, acc0,
                                                    false, false);
      acc1 = __builtin_amdgcn_wmma_f32_16x16x32_f16(false, a1, false, b, (short)0, acc1,
                                                    false, false);
    }
    int mg = (lane < 16) ? 0 : 8;
    int n  = ct * 16 + (lane & 15);
    float bb = bp[n];
#pragma unroll
    for (int v = 0; v < 8; ++v) {
      out[(size_t)(r0 + v + mg) * DIM + n]      = lrelu(acc0[v] + bb);
      out[(size_t)(r0 + v + mg + 16) * DIM + n] = lrelu(acc1[v] + bb);
    }
  }
}

// ---------------- edge remap + stable compaction ----------------
__global__ void fill_edges(float* __restrict__ eout) {
  long long i = (long long)blockIdx.x * 256 + threadIdx.x;
  if (i < 2LL * N_EDGES) eout[i] = -1.0f;
}

__global__ __launch_bounds__(256)
void edge_count(const long long* __restrict__ ei, const int* __restrict__ nmap,
                int* __restrict__ blkcnt) {
  __shared__ int sh[256];
  int t = threadIdx.x;
  long long base = (long long)blockIdx.x * EPB + t * 8;
  int c = 0;
#pragma unroll
  for (int u = 0; u < 8; ++u) {
    long long e = base + u;
    if (e < N_EDGES && nmap[ei[e]] >= 0 && nmap[ei[N_EDGES + e]] >= 0) c++;
  }
  sh[t] = c;
  __syncthreads();
  for (int s = 128; s > 0; s >>= 1) {
    if (t < s) sh[t] += sh[t + s];
    __syncthreads();
  }
  if (t == 0) blkcnt[blockIdx.x] = sh[0];
}

__global__ __launch_bounds__(1024)
void scan_blocks(const int* __restrict__ cnt, int* __restrict__ off, int nb) {
  __shared__ int sh[1024];
  __shared__ int carry_sh;
  int t = threadIdx.x;
  if (t == 0) carry_sh = 0;
  __syncthreads();
  for (int base = 0; base < nb; base += 1024) {
    int v = (base + t < nb) ? cnt[base + t] : 0;
    sh[t] = v;
    __syncthreads();
    for (int s = 1; s < 1024; s <<= 1) {
      int add = (t >= s) ? sh[t - s] : 0;
      __syncthreads();
      sh[t] += add;
      __syncthreads();
    }
    int c0 = carry_sh;
    if (base + t < nb) off[base + t] = sh[t] - v + c0;   // exclusive + carry
    __syncthreads();
    if (t == 0) carry_sh += sh[1023];
    __syncthreads();
  }
}

__global__ __launch_bounds__(256)
void edge_scatter(const long long* __restrict__ ei, const int* __restrict__ nmap,
                  const int* __restrict__ blkoff, float* __restrict__ eout) {
  __shared__ int sh[256];
  int t = threadIdx.x;
  long long base = (long long)blockIdx.x * EPB + t * 8;
  int mr[8], mc[8], c = 0;
#pragma unroll
  for (int u = 0; u < 8; ++u) {
    long long e = base + u;
    int a = -1, b = -1;
    if (e < N_EDGES) { a = nmap[ei[e]]; b = nmap[ei[N_EDGES + e]]; }
    bool ok = (a >= 0) && (b >= 0);
    mr[u] = ok ? a : -2;
    mc[u] = b;
    if (ok) c++;
  }
  sh[t] = c;
  __syncthreads();
  for (int s = 1; s < 256; s <<= 1) {
    int add = (t >= s) ? sh[t - s] : 0;
    __syncthreads();
    sh[t] += add;
    __syncthreads();
  }
  int pos = blkoff[blockIdx.x] + sh[t] - c;   // stable: thread order == edge order
#pragma unroll
  for (int u = 0; u < 8; ++u) {
    if (mr[u] >= 0) {
      eout[pos]                       = (float)mr[u];
      eout[(long long)N_EDGES + pos]  = (float)mc[u];
      pos++;
    }
  }
}

// ---------------- launcher ----------------
extern "C" void kernel_launch(void* const* d_in, const int* in_sizes, int n_in,
                              void* d_out, int out_size, void* d_ws, size_t ws_size,
                              hipStream_t stream) {
  (void)in_sizes; (void)n_in; (void)out_size; (void)ws_size;

  const float*     x     = (const float*)d_in[0];
  const long long* ei    = (const long long*)d_in[1];
  const float*     ws1   = (const float*)d_in[2];
  const float*     bs1   = (const float*)d_in[3];
  const float*     ws2   = (const float*)d_in[4];
  const float*     bs2   = (const float*)d_in[5];
  const float*     wf1   = (const float*)d_in[6];
  const float*     bf1   = (const float*)d_in[7];
  const float*     wf2   = (const float*)d_in[8];
  const float*     bf2   = (const float*)d_in[9];
  const float*     wp    = (const float*)d_in[10];
  const float*     bp    = (const float*)d_in[11];
  const float*     alpha = (const float*)d_in[12];
  const float*     beta  = (const float*)d_in[13];

  // workspace carve-out (256B aligned), total ~5.3 MB
  char* ws = (char*)d_ws;
  size_t o = 0;
  auto carve = [&](size_t bytes) { size_t r = o; o = (o + bytes + 255) & ~(size_t)255; return r; };
  float*              deg    = (float*)(ws + carve((size_t)N_NODES * 4));
  float*              sumsq  = (float*)(ws + carve(4));
  float*              scores = (float*)(ws + carve((size_t)N_NODES * 4));
  unsigned long long* keys   = (unsigned long long*)(ws + carve((size_t)M_SORT * 8));
  int*                nmap   = (int*)(ws + carve((size_t)N_NODES * 4));
  int*                topk   = (int*)(ws + carve((size_t)K_TOP * 4));
  _Float16*           ws1p   = (_Float16*)(ws + carve((size_t)DIM * DIM * 2)); // packed
  _Float16*           wf1p   = (_Float16*)(ws + carve((size_t)DIM * DH * 2));  // packed
  _Float16*           wpp    = (_Float16*)(ws + carve((size_t)DIM * DIM * 2)); // packed
  float*              csum   = (float*)(ws + carve((size_t)DIM * 4));
  int*                blkcnt = (int*)(ws + carve((size_t)NBLK_E * 4));
  int*                blkoff = (int*)(ws + carve((size_t)NBLK_E * 4));

  float* out        = (float*)d_out;
  float* out_pool   = out;                                  // [K_TOP, DIM]
  float* out_edges  = out + (size_t)K_TOP * DIM;            // [2, E]
  float* out_scores = out_edges + 2LL * N_EDGES;            // [K_TOP]

  // 1. init state
  init_kernel<<<(N_NODES + 255) / 256, 256, 0, stream>>>(deg, sumsq, nmap);
  // 2. fragment-major f16 weight packs + rank-1 column sums (L2-resident afterwards)
  pack_b<<<128, 256, 0, stream>>>(ws1, ws1p, 16, DIM);   // top half rows 0..255
  pack_b<<<64, 256, 0, stream>>>(wf1, wf1p, 8, DH);
  pack_b<<<128, 256, 0, stream>>>(wp, wpp, 16, DIM);
  csum_kernel<<<1, 256, 0, stream>>>(ws1, csum);
  // 3. degree + ||deg||^2
  degree_kernel<<<(N_EDGES + 255) / 256, 256, 0, stream>>>(ei, deg);
  sumsq_kernel<<<(N_NODES + 255) / 256, 256, 0, stream>>>(deg, sumsq);
  // 4. fused WMMA score kernel (6250 blocks x 32 rows)
  scores_kernel<<<N_NODES / ROWS, 256, 0, stream>>>(x, ws1p, bs1, ws2, bs2, wf1p, bf1,
                                                    wf2, bf2, csum, deg, sumsq,
                                                    alpha, beta, scores);
  // 5. sorted top-k: bitonic over 2^18 keys (2 MB -> L2)
  build_keys<<<M_SORT / 256, 256, 0, stream>>>(scores, keys);
  for (unsigned kk = 2; kk <= M_SORT; kk <<= 1)
    for (unsigned j = kk >> 1; j > 0; j >>= 1)
      bitonic_step<<<M_SORT / 256, 256, 0, stream>>>(keys, j, kk);
  extract_topk<<<(K_TOP + 255) / 256, 256, 0, stream>>>(keys, scores, topk, nmap, out_scores);
  // 6. pooled WMMA GEMM with row gather (3125 blocks x 32 rows)
  pool_kernel<<<K_TOP / ROWS, 256, 0, stream>>>(x, topk, wpp, bp, out_pool);
  // 7. edge remap + stable compaction
  fill_edges<<<(int)((2LL * N_EDGES + 255) / 256), 256, 0, stream>>>(out_edges);
  edge_count<<<NBLK_E, 256, 0, stream>>>(ei, nmap, blkcnt);
  scan_blocks<<<1, 1024, 0, stream>>>(blkcnt, blkoff, NBLK_E);
  edge_scatter<<<NBLK_E, 256, 0, stream>>>(ei, nmap, blkoff, out_edges);
}
